// CausalSelfAttention_22016002359635
// MI455X (gfx1250) — compile-verified, matrix operand order checked by reading, not run on hardware
//
#include <hip/hip_runtime.h>
#include <hip/hip_bf16.h>
#include <stdint.h>

#define B_   64
#define T_   512
#define D_   1024
#define DK_  16
#define BT_  (B_*T_)
#define NEG_INF_REF (-1e24f)

typedef __attribute__((ext_vector_type(16))) __bf16 v16bf;
typedef __attribute__((ext_vector_type(8)))  float  v8f;

#if __has_builtin(__builtin_amdgcn_wmma_f32_16x16x4_f32)
typedef __attribute__((ext_vector_type(2)))  float  v2f;
#define HAVE_WMMA_F32 1
#endif

#if defined(HAVE_WMMA_F32) && \
    __has_builtin(__builtin_amdgcn_tensor_load_to_lds) && \
    __has_builtin(__builtin_amdgcn_s_wait_tensorcnt)
#define HAVE_TDM 1
typedef unsigned int u32x4 __attribute__((ext_vector_type(4)));
typedef int          i32x4 __attribute__((ext_vector_type(4)));
typedef int          i32x8 __attribute__((ext_vector_type(8)));
#endif

// round-to-nearest-even f32 -> bf16 (inputs are finite)
__device__ __forceinline__ uint16_t f2bf(float f) {
  uint32_t u = __float_as_uint(f);
  uint32_t r = u + 0x7FFFu + ((u >> 16) & 1u);
  return (uint16_t)(r >> 16);
}

union Frag16 {          // 16 bf16 = 8 VGPRs, loaded as 2x b128
  uint4  q[2];
  v16bf  v;
};

union Pack8 {           // 8 bf16 packed for one b128 store
  uint16_t h[8];
  uint4    u;
};

#ifdef HAVE_TDM
// 2D TDM load: tile_h rows x tile_w f32 elems, row stride `row_stride` elems,
// LDS-padded by `pad_amount_code`+1 dwords every 2^(pad_interval_code+1) dwords.
__device__ __forceinline__ void tdm_load_2d_f32(
    uint32_t lds_off, const void* gaddr,
    uint32_t tile_w, uint32_t tile_h, uint32_t row_stride,
    uint32_t pad_interval_code, uint32_t pad_amount_code)
{
  const uint64_t ga = (uint64_t)(uintptr_t)gaddr;
  u32x4 g0;
  g0[0] = 1u;                                             // count=1, user desc
  g0[1] = lds_off;                                        // lds_addr (bytes)
  g0[2] = (uint32_t)ga;                                   // global_addr[31:0]
  g0[3] = (uint32_t)((ga >> 32) & 0x01FFFFFFu) | (2u << 30); // addr[56:32], type=2
  i32x8 g1;
  g1[0] = (int)((2u << 16)                 // data_size = 4B
              | (1u << 20)                 // pad_enable
              | (pad_interval_code << 22)
              | (pad_amount_code  << 25));
  g1[1] = (int)((row_stride & 0xFFFFu) << 16);            // tensor_dim0[15:0]
  g1[2] = (int)(((row_stride >> 16) & 0xFFFFu)            // tensor_dim0[31:16]
              | (tile_h << 16));                          // tensor_dim1[15:0]
  g1[3] = (int)(tile_w << 16);                            // tile_dim0
  g1[4] = (int)tile_h;                                    // tile_dim1 (tile_dim2=0)
  g1[5] = (int)row_stride;                                // tensor_dim0_stride[31:0]
  g1[6] = 0;
  g1[7] = 0;
  i32x4 z4 = {0, 0, 0, 0};
#if __clang_major__ >= 23
  i32x8 z8 = {0, 0, 0, 0, 0, 0, 0, 0};
  __builtin_amdgcn_tensor_load_to_lds(g0, g1, z4, z4, z8, 0);
#else
  __builtin_amdgcn_tensor_load_to_lds(g0, g1, z4, z4, 0);
#endif
}
#endif

// ---------------------------------------------------------------------------
// Kernel 1: q = x@Wq + bq ; k = mask ? -1e24 : x@Wk + bk     (f32)
// 128 threads = 4 waves; each wave owns a 16-row tile (block covers 64 rows).
// TDM path: per-wave double-buffered TENSOR_LOAD_TO_LDS of x tiles (16x64 f32,
// LDS row stride 65 dwords -> conflict-free), f32 WMMA 16x16x4 over K.
// ---------------------------------------------------------------------------
__global__ __launch_bounds__(128) void qk_kernel(
    const float* __restrict__ x, const uint8_t* __restrict__ pmask,
    const float* __restrict__ Wq, const float* __restrict__ bq,
    const float* __restrict__ Wk, const float* __restrict__ bk,
    float* __restrict__ qo, float* __restrict__ ko)
{
#ifdef HAVE_WMMA_F32
  const int wid  = threadIdx.x >> 5;
  const int lane = threadIdx.x & 31;
  const int r0   = blockIdx.x * 64 + wid * 16;
  const int m    = lane & 15;      // A: M row / B: N col
  const int hi   = lane >> 4;      // K-half selector
  v8f accq = {0,0,0,0,0,0,0,0};
  v8f acck = {0,0,0,0,0,0,0,0};

#ifdef HAVE_TDM
  // [wave][buf][16 rows * 65 dwords (64 data + 1 pad)]
  __shared__ float xtile[4][2][16 * 65];
  const uint32_t lds_base = (uint32_t)(uintptr_t)&xtile[0][0][0];
  const uint32_t my0 = lds_base + (uint32_t)(wid * 2) * (16 * 65 * 4);
  const uint32_t my1 = my0 + (16 * 65 * 4);
  const float* xrow0 = x + (size_t)r0 * D_;

  // prologue: fetch k0 = 0 into buf 0     (pad: every 64 dwords, +1 dword)
  tdm_load_2d_f32(my0, xrow0, 64, 16, D_, /*interval 2^6=64dw*/5, /*1dw*/0);
  int buf = 0;
  for (int k0 = 0; k0 < D_; k0 += 64) {
    const bool more = (k0 + 64) < D_;
    if (more) {
      tdm_load_2d_f32(buf ? my0 : my1, xrow0 + k0 + 64, 64, 16, D_, 5, 0);
      __builtin_amdgcn_s_wait_tensorcnt(1);   // oldest (current buf) done
    } else {
      __builtin_amdgcn_s_wait_tensorcnt(0);
    }
    const float* xa = &xtile[wid][buf][0];
    #pragma unroll
    for (int kc = 0; kc < 64; kc += 4) {
      const int kk = kc + 2 * hi;
      const int gk = k0 + kk;
      v2f a;   a[0]  = xa[m * 65 + kk];     a[1]  = xa[m * 65 + kk + 1];
      v2f bq2; bq2[0]= Wq[gk*DK_ + m];      bq2[1]= Wq[(gk+1)*DK_ + m];
      v2f bk2; bk2[0]= Wk[gk*DK_ + m];      bk2[1]= Wk[(gk+1)*DK_ + m];
      accq = __builtin_amdgcn_wmma_f32_16x16x4_f32(false, a, false, bq2,
                                                   (short)0, accq, false, false);
      acck = __builtin_amdgcn_wmma_f32_16x16x4_f32(false, a, false, bk2,
                                                   (short)0, acck, false, false);
    }
    buf ^= 1;
  }
#else
  const float* xrow = x + (size_t)(r0 + m) * D_;
  for (int kc = 0; kc < D_; kc += 4) {
    const int kk = kc + 2 * hi;
    v2f a;   a[0]  = xrow[kk];            a[1]  = xrow[kk + 1];
    v2f bq2; bq2[0]= Wq[kk*DK_ + m];      bq2[1]= Wq[(kk+1)*DK_ + m];
    v2f bk2; bk2[0]= Wk[kk*DK_ + m];      bk2[1]= Wk[(kk+1)*DK_ + m];
    accq = __builtin_amdgcn_wmma_f32_16x16x4_f32(false, a, false, bq2,
                                                 (short)0, accq, false, false);
    acck = __builtin_amdgcn_wmma_f32_16x16x4_f32(false, a, false, bk2,
                                                 (short)0, acck, false, false);
  }
#endif
  const float bqn = bq[m], bkn = bk[m];
  #pragma unroll
  for (int i = 0; i < 8; ++i) {
    const int row = r0 + i + 8 * hi;       // D layout: M = i + 8*hi
    qo[row*DK_ + m] = accq[i] + bqn;
    ko[row*DK_ + m] = pmask[row] ? NEG_INF_REF : (acck[i] + bkn);
  }
#else
  // VALU fallback
  const int n = threadIdx.x & 15;
  for (int rr = threadIdx.x >> 4; rr < 64; rr += 8) {
    const int row = blockIdx.x * 64 + rr;
    const float* xr = x + (size_t)row * D_;
    float aq = 0.f, ak = 0.f;
    for (int kk = 0; kk < D_; ++kk) {
      const float xv = xr[kk];
      aq += xv * Wq[kk*DK_ + n];
      ak += xv * Wk[kk*DK_ + n];
    }
    qo[row*DK_ + n] = aq + bq[n];
    ko[row*DK_ + n] = pmask[row] ? NEG_INF_REF : (ak + bk[n]);
  }
#endif
}

// ---------------------------------------------------------------------------
// Kernel 2: vT[b][d][t] = bf16( (x @ Wv + bv)[b][t][d] )
// 128x128 macro-tile per block, 8 waves, K-step 32, bf16 WMMA 16x16x32.
// Register double-buffered staging: next tile's global loads are issued right
// after the first barrier so they overlap the 8 WMMAs of the current step.
// ---------------------------------------------------------------------------
__global__ __launch_bounds__(256) void v_kernel(
    const float* __restrict__ x, const float* __restrict__ Wv,
    const float* __restrict__ bv, uint16_t* __restrict__ vT)
{
  __shared__ __align__(16) uint16_t As[128][40];  // [m][k], 80B stride (conflict-free)
  __shared__ __align__(16) uint16_t Bs[128][40];  // [n][k] = Wv^T tile

  const int tid  = threadIdx.x;
  const int wid  = tid >> 5;
  const int lane = tid & 31;
  const int row0 = blockIdx.y * 128;   // token rows
  const int col0 = blockIdx.x * 128;   // d columns

  v8f acc[8];
  #pragma unroll
  for (int j = 0; j < 8; ++j) acc[j] = (v8f){0,0,0,0,0,0,0,0};

  const int am = wid * 16 + (lane & 15);
  const int ak = (lane >> 4) * 16;

  const int ar = tid >> 1, ah = (tid & 1) * 16;          // A staging coords
  const int bk_ = tid >> 3, bn = (tid & 7) * 16;         // B staging coords
  const float* aptr = x  + (size_t)(row0 + ar) * D_ + ah;
  const float* bptr = Wv + (size_t)bk_ * D_ + col0 + bn;

  float4 a0, a1, a2, a3, b0, b1, b2, b3;
  // prologue: load k0 = 0
  {
    const float4* ap = (const float4*)aptr;
    a0 = ap[0]; a1 = ap[1]; a2 = ap[2]; a3 = ap[3];
    const float4* bp = (const float4*)bptr;
    b0 = bp[0]; b1 = bp[1]; b2 = bp[2]; b3 = bp[3];
  }

  for (int k0 = 0; k0 < D_; k0 += 32) {
    // ---- commit staged registers to LDS (f32 -> bf16)
    {
      const float av[16] = {a0.x,a0.y,a0.z,a0.w, a1.x,a1.y,a1.z,a1.w,
                            a2.x,a2.y,a2.z,a2.w, a3.x,a3.y,a3.z,a3.w};
      Pack8 p0, p1;
      #pragma unroll
      for (int u = 0; u < 8; ++u) { p0.h[u] = f2bf(av[u]); p1.h[u] = f2bf(av[u+8]); }
      *(uint4*)&As[ar][ah]     = p0.u;
      *(uint4*)&As[ar][ah + 8] = p1.u;

      const float bw[16] = {b0.x,b0.y,b0.z,b0.w, b1.x,b1.y,b1.z,b1.w,
                            b2.x,b2.y,b2.z,b2.w, b3.x,b3.y,b3.z,b3.w};
      #pragma unroll
      for (int u = 0; u < 16; ++u) Bs[bn + u][bk_] = f2bf(bw[u]);   // transpose
    }
    __syncthreads();

    // ---- issue next tile's global loads (overlap with WMMAs below)
    if (k0 + 32 < D_) {
      const float4* ap = (const float4*)(aptr + k0 + 32);
      a0 = ap[0]; a1 = ap[1]; a2 = ap[2]; a3 = ap[3];
      const float4* bp = (const float4*)(bptr + (size_t)(k0 + 32) * D_);
      b0 = bp[0]; b1 = bp[1]; b2 = bp[2]; b3 = bp[3];
      __builtin_prefetch(aptr + k0 + 64, 0, 0);
    }

    // ---- 8 WMMAs on current tile
    Frag16 a;
    a.q[0] = *(const uint4*)&As[am][ak];
    a.q[1] = *(const uint4*)&As[am][ak + 8];
    #pragma unroll
    for (int j = 0; j < 8; ++j) {
      Frag16 b;
      const int jn = j * 16 + (lane & 15);
      b.q[0] = *(const uint4*)&Bs[jn][ak];
      b.q[1] = *(const uint4*)&Bs[jn][ak + 8];
      acc[j] = __builtin_amdgcn_wmma_f32_16x16x32_bf16(
          false, a.v, false, b.v, (short)0, acc[j], false, false);
    }
    __syncthreads();
  }

  // ---- epilogue: +bv, bf16, store transposed vT[b][d][t]
  const int hi = lane >> 4;
  const int trow = row0 + wid * 16 + 8 * hi;   // first of 8 consecutive tokens
  const int bb = trow >> 9;
  const int t  = trow & (T_ - 1);
  #pragma unroll
  for (int j = 0; j < 8; ++j) {
    const int d = col0 + j * 16 + (lane & 15);
    const float bvn = bv[d];
    Pack8 p;
    #pragma unroll
    for (int i = 0; i < 8; ++i) p.h[i] = f2bf(acc[j][i] + bvn);
    *(uint4*)&vT[((size_t)bb * D_ + d) * T_ + t] = p.u;
  }
}

// ---------------------------------------------------------------------------
// Kernel 3: per (batch, 16-query-row tile): scores (f32, *4, causal) ->
// softmax -> P(bf16) @ vT  with bf16 WMMA.  128 threads = 4 waves.
// ---------------------------------------------------------------------------
__global__ __launch_bounds__(128) void attn_kernel(
    const float* __restrict__ qi, const float* __restrict__ ki,
    const uint16_t* __restrict__ vT, float* __restrict__ out)
{
  __shared__ float S[16][512];
  __shared__ __align__(16) uint16_t P[16][520];   // 1040B stride: 16B aligned, conflict-free
  __shared__ float red[16][8];
  __shared__ float rowmax[16];
  __shared__ float rowinv[16];

  const int tid  = threadIdx.x;
  const int wid  = tid >> 5;
  const int lane = tid & 31;
  const int b    = blockIdx.x >> 5;
  const int t0   = (blockIdx.x & 31) * 16;
  const int Sc   = t0 + 16;                // causal column count
  const int ScPad= (Sc + 31) & ~31;        // pad K to multiple of 32 (<=512)
  const int ntiles = Sc >> 4;

  const float* qb = qi + ((size_t)b * T_ + t0) * DK_;
  const float* kb = ki + (size_t)b * T_ * DK_;

  // ---------------- scores = (q @ k^T) * 4, causal mask, into S ----------
#ifdef HAVE_WMMA_F32
  {
    const int m  = lane & 15;
    const int hi = lane >> 4;
    for (int st = wid; st < ntiles; st += 4) {
      v8f acc = (v8f){0,0,0,0,0,0,0,0};
      #pragma unroll
      for (int kc = 0; kc < DK_; kc += 4) {
        const int kk = kc + 2 * hi;
        v2f a;  a[0]  = qb[m*DK_ + kk];               a[1]  = qb[m*DK_ + kk + 1];
        v2f bv2;bv2[0]= kb[(st*16 + m)*DK_ + kk];     bv2[1]= kb[(st*16 + m)*DK_ + kk + 1];
        acc = __builtin_amdgcn_wmma_f32_16x16x4_f32(false, a, false, bv2,
                                                    (short)0, acc, false, false);
      }
      #pragma unroll
      for (int i = 0; i < 8; ++i) {
        const int mr = i + 8 * hi;
        const int s  = st * 16 + m;
        float v = acc[i] * 4.0f;                 // * sqrt(dk)
        if (s > t0 + mr) v = NEG_INF_REF;        // causal
        S[mr][s] = v;
      }
    }
  }
#else
  for (int idx = tid; idx < 16 * Sc; idx += 128) {
    const int mr = idx & 15;
    const int s  = idx >> 4;
    float dot = 0.f;
    #pragma unroll
    for (int kk = 0; kk < DK_; ++kk)
      dot += qb[mr*DK_ + kk] * kb[s*DK_ + kk];
    float v = dot * 4.0f;
    if (s > t0 + mr) v = NEG_INF_REF;
    S[mr][s] = v;
  }
#endif
  __syncthreads();

  // ---------------- softmax over s (two-pass), P = bf16(exp), unnormalized
  const int r  = tid >> 3;
  const int c8 = tid & 7;
  float mx = -3.4e38f;
  for (int s = c8; s < Sc; s += 8) mx = fmaxf(mx, S[r][s]);
  red[r][c8] = mx;
  __syncthreads();
  if (tid < 16) {
    float v = red[tid][0];
    #pragma unroll
    for (int u = 1; u < 8; ++u) v = fmaxf(v, red[tid][u]);
    rowmax[tid] = v;
  }
  __syncthreads();
  {
    const float rmx = rowmax[r];
    float sm = 0.f;
    for (int s = c8; s < Sc; s += 8) {
      const float e = __expf(S[r][s] - rmx);
      P[r][s] = f2bf(e);
      sm += e;
    }
    for (int s = Sc + c8; s < ScPad; s += 8) P[r][s] = 0;  // zero K padding
    red[r][c8] = sm;
  }
  __syncthreads();
  if (tid < 16) {
    float v = 0.f;
    #pragma unroll
    for (int u = 0; u < 8; ++u) v += red[tid][u];
    rowinv[tid] = 1.0f / v;
  }
  __syncthreads();

  // ---------------- out[t0..t0+15][:] = (P @ v) * rowinv ----------------
  const uint16_t* vb = vT + (size_t)b * D_ * T_;
  const int m2  = lane & 15;
  const int hi2 = lane >> 4;
  for (int nt = wid; nt < D_ / 16; nt += 4) {
    v8f acc = (v8f){0,0,0,0,0,0,0,0};
    for (int kb2 = 0; kb2 < ScPad; kb2 += 32) {
      const int ks = kb2 + hi2 * 16;
      Frag16 a, bf;
      a.q[0] = *(const uint4*)&P[m2][ks];
      a.q[1] = *(const uint4*)&P[m2][ks + 8];
      const uint16_t* bp = vb + (size_t)(nt * 16 + m2) * T_ + ks;
      bf.q[0] = *(const uint4*)bp;
      bf.q[1] = *(const uint4*)(bp + 8);
      acc = __builtin_amdgcn_wmma_f32_16x16x32_bf16(
          false, a.v, false, bf.v, (short)0, acc, false, false);
    }
    #pragma unroll
    for (int i = 0; i < 8; ++i) {
      const int mr = i + 8 * hi2;
      out[((size_t)b * T_ + t0 + mr) * D_ + nt * 16 + m2] = acc[i] * rowinv[mr];
    }
  }
}

// ---------------------------------------------------------------------------
extern "C" void kernel_launch(void* const* d_in, const int* in_sizes, int n_in,
                              void* d_out, int out_size, void* d_ws, size_t ws_size,
                              hipStream_t stream) {
  const float*   x     = (const float*)  d_in[0];
  const uint8_t* pmask = (const uint8_t*)d_in[1];   // bool [B,T,1]
  const float*   Wq    = (const float*)  d_in[2];
  const float*   bq    = (const float*)  d_in[3];
  const float*   Wk    = (const float*)  d_in[4];
  const float*   bk    = (const float*)  d_in[5];
  const float*   Wv    = (const float*)  d_in[6];
  const float*   bv    = (const float*)  d_in[7];
  float*         out   = (float*)d_out;

  char* ws = (char*)d_ws;
  float*    qbuf = (float*)ws;                               // BT*16*4 = 2 MiB
  float*    kbuf = (float*)(ws + (size_t)2  * 1024 * 1024);  // 2 MiB
  uint16_t* vT   = (uint16_t*)(ws + (size_t)4 * 1024 * 1024);// BT*D*2 = 64 MiB

  qk_kernel<<<BT_ / 64, 128, 0, stream>>>(x, pmask, Wq, bq, Wk, bk, qbuf, kbuf);
  v_kernel<<<dim3(D_ / 128, BT_ / 128), 256, 0, stream>>>(x, Wv, bv, vT);
  attn_kernel<<<B_ * (T_ / 16), 128, 0, stream>>>(qbuf, kbuf, vT, out);
}
